// GNNVectorContinuousMultiHeadAttention_66529043415205
// MI455X (gfx1250) — matrix operand order repaired and checked by reading, compiled, not run on hardware
//
#include <hip/hip_runtime.h>
#include <hip/hip_bf16.h>
#include <math.h>

// ---------------------------------------------------------------------------
// Problem constants (B, LQ, LK, D, H) = (4, 192, 192, 512, 8), d_q = 64.
// ---------------------------------------------------------------------------
constexpr int B_  = 4;
constexpr int LQ  = 192;
constexpr int LK  = 192;
constexpr int DM  = 512;
constexpr int ROWS = B_ * LQ;          // 768 flattened (b, q) / (b, k) rows

typedef __bf16 v16bf __attribute__((ext_vector_type(16)));
typedef float  v8f   __attribute__((ext_vector_type(8)));
typedef unsigned int u32x4 __attribute__((ext_vector_type(4)));
typedef int          i32x4 __attribute__((ext_vector_type(4)));
typedef int          i32x8 __attribute__((ext_vector_type(8)));

// Tensor Data Mover availability / arity detection (see CDNA5_HIP.md):
//  - ROCm 7.2 (clang-22): 5-arg builtin, no TDM header.
//  - amdgpu-toolchain (clang-23 + therock-10.0): 6-arg builtin, ships the
//    gfx1250 TDM header. Use the header's presence as the arity switch.
#if __has_builtin(__builtin_amdgcn_tensor_load_to_lds) && \
    __has_builtin(__builtin_amdgcn_s_wait_tensorcnt)
#define HAVE_TDM 1
#if __has_include(<hip/amd_detail/amd_gfx1250_TDM.h>)
#define TDM_ARITY6 1
#endif
#endif

static __device__ __forceinline__ v8f wmma_bf16(v16bf a, v16bf b, v8f c) {
  // D(16x16,f32) = A(16x32,bf16) * B(32x16,bf16) + C
  return __builtin_amdgcn_wmma_f32_16x16x32_bf16(false, a, false, b,
                                                 (short)0, c, false, false);
}

static __device__ __forceinline__ unsigned short bf_bits(float f) {
  __bf16 h = (__bf16)f;
  return __builtin_bit_cast(unsigned short, h);
}
static __device__ __forceinline__ float bits_f(unsigned short u) {
  return (float)__builtin_bit_cast(__bf16, u);
}
static __device__ __forceinline__ __bf16 bits_bf(unsigned short u) {
  return __builtin_bit_cast(__bf16, u);
}

struct __align__(8) US4 { unsigned short x, y, z, w; };

// ---------------------------------------------------------------------------
// Kernel 1: fused input projections.
//   qb = bf16(q @ w_q^T + b_q)        [768 x 512]
//   kb = bf16(k @ w_k^T)              [768 x 512]
//   db = bf16(k @ w_v^T - k @ w_k^T)  [768 x 512]  (vproj - kproj)
// One 16x16 output tile per wave; K loop over 512 in steps of 32.
// ---------------------------------------------------------------------------
__global__ void __launch_bounds__(256)
proj_kernel(const float* __restrict__ q, const float* __restrict__ k,
            const float* __restrict__ w_q, const float* __restrict__ b_q,
            const float* __restrict__ w_k, const float* __restrict__ w_v,
            unsigned short* __restrict__ qb, unsigned short* __restrict__ kb,
            unsigned short* __restrict__ db)
{
  const int lane = threadIdx.x & 31;
  const int wave = threadIdx.x >> 5;
  const int tile = blockIdx.x * 8 + wave;      // 48 row-tiles x 32 col-tiles
  const int rt = tile >> 5;
  const int ct = tile & 31;
  const int row = rt * 16 + (lane & 15);
  const int col = ct * 16 + (lane & 15);

  v8f aq = {}, ak = {}, av = {};
  for (int kk = 0; kk < DM; kk += 32) {
    const int ka = kk + (lane >> 4) * 16;
    const int kbB = kk + (lane >> 4) * 16;
    v16bf Aq, Ak, Bq, Bk, Bv;
#pragma unroll
    for (int j = 0; j < 8; ++j) {
      Aq[j]     = (__bf16)q[row * DM + ka + j];
      Aq[j + 8] = (__bf16)q[row * DM + ka + 8 + j];
      Ak[j]     = (__bf16)k[row * DM + ka + j];
      Ak[j + 8] = (__bf16)k[row * DM + ka + 8 + j];
    }
#pragma unroll
    for (int e = 0; e < 16; ++e) {
      Bq[e] = (__bf16)w_q[col * DM + kbB + e];
      Bk[e] = (__bf16)w_k[col * DM + kbB + e];
      Bv[e] = (__bf16)w_v[col * DM + kbB + e];
    }
    aq = wmma_bf16(Aq, Bq, aq);
    ak = wmma_bf16(Ak, Bk, ak);
    av = wmma_bf16(Ak, Bv, av);
  }

  const float bias = b_q[col];
#pragma unroll
  for (int r = 0; r < 8; ++r) {
    const int orow = rt * 16 + r + 8 * (lane >> 4);   // C layout: M = r + 8*(l>>4)
    qb[orow * DM + col] = bf_bits(aq[r] + bias);
    kb[orow * DM + col] = bf_bits(ak[r]);
    db[orow * DM + col] = bf_bits(av[r] - ak[r]);
  }
}

// ---------------------------------------------------------------------------
// Kernel 2: attention core. One workgroup (256 thr = 8 waves) per (b, q).
// edge_mask[b,q] (192x512 fp32, 384 KB HBM) is streamed exactly ONCE —
// via the Tensor Data Mover into a double-buffered LDS staging area when
// available — and fused into  s_key[k][d] = bf16(edge + kproj)  (198 KB LDS).
// Logits via WMMA against a block-diagonal 16x512 Q matrix, per-head wave32
// softmax, then in-place s_key += (vproj - kproj) and the output WMMA GEMM.
// ---------------------------------------------------------------------------
constexpr int KST = 516;   // padded bf16 row stride: 258 dwords = 2 mod 64
                           // -> column walks (output GEMM B frag) hit 16
                           //    distinct LDS banks; 8-byte aligned rows.
constexpr int LST = 208;   // logits row stride (floats)
constexpr int PST = 224;   // probs row stride (bf16)
#ifdef HAVE_TDM
constexpr int CH_ROWS = 8;             // k-rows per TDM chunk (16 KB fp32)
constexpr int NCH = LK / CH_ROWS;      // 24 chunks
#endif

__global__ void __launch_bounds__(256)
attn_kernel(const float* __restrict__ edge, const unsigned char* __restrict__ pmask,
            const unsigned short* __restrict__ qb, const unsigned short* __restrict__ kb,
            const unsigned short* __restrict__ db, unsigned short* __restrict__ attno)
{
  __shared__ unsigned short s_key[LK * KST];   // 198,144 B : key_eff -> val_eff
  __shared__ unsigned short s_q[16 * DM];      //  16,384 B : block-diag query
  __shared__ float          s_logit[16 * LST]; //  13,312 B : logits [head][k]
  __shared__ unsigned short s_p[16 * PST];     //   7,168 B : softmax probs bf16
#ifdef HAVE_TDM
  __shared__ float s_stage[2][CH_ROWS * DM];   //  32,768 B : TDM fp32 staging
#endif

  const int bq   = blockIdx.x;       // flat (b*LQ + q)
  const int b    = bq / LQ;
  const int tid  = threadIdx.x;
  const int lane = tid & 31;
  const int wave = tid >> 5;

  const float*          eg  = edge + (size_t)bq * (LK * DM);
  const unsigned short* kpb = kb + (size_t)b * LK * DM;

#ifdef HAVE_TDM
  // ---- Phase 1 (TDM): DMA fp32 edge chunks -> staging, convert+fuse -------
  auto tdm_issue = [&](int c, int buf) {
    const size_t ga = (size_t)(const void*)eg + (size_t)c * CH_ROWS * DM * 4;
    const unsigned lds_addr = (unsigned)(size_t)(const void*)&s_stage[buf][0];
    u32x4 g0;
    g0[0] = 1u;                                        // count=1 (valid D#)
    g0[1] = lds_addr;                                  // lds_addr [63:32]
    g0[2] = (unsigned)(ga & 0xFFFFFFFFu);              // global_addr lo
    g0[3] = (unsigned)((ga >> 32) & 0x01FFFFFFu)       // global_addr hi
          | (2u << 30);                                // type=2 ("image")
    i32x8 g1;
    g1[0] = 0x20000;                 // workgroup_mask=0, data_size=2 (4 B)
    g1[1] = (int)(512u << 16);       // tensor_dim0 = 512 (bits [79:48] lo)
    g1[2] = (int)((unsigned)CH_ROWS << 16);  // tensor_dim1 = 8
    g1[3] = (int)(512u << 16);       // tile_dim0 = 512 (bits [127:112])
    g1[4] = CH_ROWS;                 // tile_dim1 = 8; tile_dim2 = 0
    g1[5] = 512;                     // tensor_dim0_stride = 512
    g1[6] = 0;
    g1[7] = 0;
    const i32x4 z4 = {0, 0, 0, 0};
#ifdef TDM_ARITY6
    const i32x8 z8 = {0, 0, 0, 0, 0, 0, 0, 0};
    __builtin_amdgcn_tensor_load_to_lds(g0, g1, z4, z4, z8, 0);
#else
    __builtin_amdgcn_tensor_load_to_lds(g0, g1, z4, z4, 0);
#endif
  };

  if (wave == 0) {               // one wave drives the DMA pipeline
    tdm_issue(0, 0);
    tdm_issue(1, 1);
  }
#endif

  // Block-diagonal query matrix: row h holds head-h slice of qproj, else 0.
  // (overlaps with the first DMA chunk in flight)
  {
    const unsigned short* qrow = qb + (size_t)bq * DM;
    for (int idx = tid; idx < 16 * DM; idx += 256) {
      const int h = idx >> 9;
      const int d = idx & (DM - 1);
      s_q[idx] = (h == (d >> 6)) ? qrow[d] : (unsigned short)0;
    }
    for (int idx = tid; idx < 16 * PST; idx += 256) s_p[idx] = 0;
  }

#ifdef HAVE_TDM
  for (int c = 0; c < NCH; ++c) {
    if (wave == 0) {
      // SOPP immediate: builtin requires a compile-time constant operand.
      if (c + 1 < NCH) __builtin_amdgcn_s_wait_tensorcnt(1);  // chunk c done,
      else             __builtin_amdgcn_s_wait_tensorcnt(0);  // all done
    }
    __syncthreads();
    const float* stg = &s_stage[c & 1][0];
#pragma unroll
    for (int i = 0; i < (CH_ROWS * DM) / 1024; ++i) {
      const int idx = tid * 4 + i * 1024;
      const int r = idx >> 9;
      const int d = idx & (DM - 1);
      const int kr = c * CH_ROWS + r;
      const float4 e4 = *reinterpret_cast<const float4*>(stg + idx);
      const US4 kp = *reinterpret_cast<const US4*>(kpb + kr * DM + d);
      US4 o;
      o.x = bf_bits(e4.x + bits_f(kp.x));
      o.y = bf_bits(e4.y + bits_f(kp.y));
      o.z = bf_bits(e4.z + bits_f(kp.z));
      o.w = bf_bits(e4.w + bits_f(kp.w));
      *reinterpret_cast<US4*>(&s_key[kr * KST + d]) = o;
    }
    __syncthreads();               // staging buffer (c&1) now free
    if (wave == 0 && c + 2 < NCH) tdm_issue(c + 2, c & 1);
  }
#else
  // ---- Phase 1 (fallback): synchronous load + convert + fuse --------------
  for (int idx = tid * 4; idx < LK * DM; idx += 256 * 4) {
    const int kr = idx >> 9;
    const int d  = idx & (DM - 1);
    const float4 e4 = *reinterpret_cast<const float4*>(eg + idx);
    const US4 kp = *reinterpret_cast<const US4*>(kpb + kr * DM + d);
    US4 o;
    o.x = bf_bits(e4.x + bits_f(kp.x));
    o.y = bf_bits(e4.y + bits_f(kp.y));
    o.z = bf_bits(e4.z + bits_f(kp.z));
    o.w = bf_bits(e4.w + bits_f(kp.w));
    *reinterpret_cast<US4*>(&s_key[kr * KST + d]) = o;
  }
  __syncthreads();
#endif

  // ---- Phase 2: logits L[k, h] = sum_d key_eff[k,d] * Qdiag[h,d] ----------
  for (int kt = wave; kt < LK / 16; kt += 8) {
    v8f acc = {};
    const int krow = kt * 16 + (lane & 15);
    const int col  = lane & 15;               // head column
    for (int dd = 0; dd < DM; dd += 32) {
      const int ka  = dd + (lane >> 4) * 16;
      const int kbb = dd + (lane >> 4) * 16;
      v16bf A, Bf;
#pragma unroll
      for (int j = 0; j < 8; ++j) {
        A[j]     = bits_bf(s_key[krow * KST + ka + j]);
        A[j + 8] = bits_bf(s_key[krow * KST + ka + 8 + j]);
      }
#pragma unroll
      for (int e = 0; e < 16; ++e) Bf[e] = bits_bf(s_q[col * DM + kbb + e]);
      acc = wmma_bf16(A, Bf, acc);
    }
#pragma unroll
    for (int r = 0; r < 8; ++r) {
      const int kidx = kt * 16 + r + 8 * (lane >> 4);
      s_logit[col * LST + kidx] = acc[r] * 0.125f;   // 1/sqrt(d_q), d_q=64
    }
  }
  __syncthreads();

  // ---- Phase 3: masked softmax over k, one head per wave ------------------
  if (wave < 8) {
    const int h = wave;
    const unsigned char* pm = pmask + (size_t)bq * LK;
    float vals[6];
    float m = -__builtin_inff();
#pragma unroll
    for (int i = 0; i < 6; ++i) {
      const int kk2 = lane + i * 32;
      float L = s_logit[h * LST + kk2];
      if (pm[kk2]) L = -__builtin_inff();
      vals[i] = L;
      m = fmaxf(m, L);
    }
#pragma unroll
    for (int off = 16; off > 0; off >>= 1) m = fmaxf(m, __shfl_xor(m, off, 32));
    float sum = 0.f;
#pragma unroll
    for (int i = 0; i < 6; ++i) { vals[i] = __expf(vals[i] - m); sum += vals[i]; }
#pragma unroll
    for (int off = 16; off > 0; off >>= 1) sum += __shfl_xor(sum, off, 32);
    const float inv = 1.0f / sum;
#pragma unroll
    for (int i = 0; i < 6; ++i) s_p[h * PST + lane + i * 32] = bf_bits(vals[i] * inv);
  }
  __syncthreads();

  // ---- Phase 4: val_eff = key_eff + (vproj - kproj) in place --------------
  {
    const unsigned short* dp = db + (size_t)b * LK * DM;
    for (int idx = tid * 4; idx < LK * DM; idx += 256 * 4) {
      const int kr = idx >> 9;
      const int d  = idx & (DM - 1);
      US4* ptr = reinterpret_cast<US4*>(&s_key[kr * KST + d]);
      US4 cur = *ptr;
      const US4 dd = *reinterpret_cast<const US4*>(dp + kr * DM + d);
      cur.x = bf_bits(bits_f(cur.x) + bits_f(dd.x));
      cur.y = bf_bits(bits_f(cur.y) + bits_f(dd.y));
      cur.z = bf_bits(bits_f(cur.z) + bits_f(dd.z));
      cur.w = bf_bits(bits_f(cur.w) + bits_f(dd.w));
      *ptr = cur;
    }
  }
  __syncthreads();

  // ---- Phase 5: O[h, d] = sum_k P[h,k] * val_eff[k,d]; pick row head(d) ---
  unsigned short* arow = attno + (size_t)bq * DM;
  for (int dt = wave; dt < DM / 16; dt += 8) {
    v8f acc = {};
    const int hrow = lane & 15;
    const int dcol = dt * 16 + (lane & 15);
    for (int kk = 0; kk < LK; kk += 32) {
      const int ka  = kk + (lane >> 4) * 16;
      const int kbb = kk + (lane >> 4) * 16;
      v16bf A, Bf;
#pragma unroll
      for (int j = 0; j < 8; ++j) {
        A[j]     = bits_bf(s_p[hrow * PST + ka + j]);
        A[j + 8] = bits_bf(s_p[hrow * PST + ka + 8 + j]);
      }
#pragma unroll
      for (int e = 0; e < 16; ++e) Bf[e] = bits_bf(s_key[(kbb + e) * KST + dcol]);
      acc = wmma_bf16(A, Bf, acc);
    }
    const int head = dcol >> 6;                // d_q = 64
#pragma unroll
    for (int r = 0; r < 8; ++r) {
      const int M = r + 8 * (lane >> 4);
      if (M == head) arow[dcol] = bf_bits(acc[r]);
    }
  }
}

// ---------------------------------------------------------------------------
// Kernel 3: final projection  out = attno @ w_h^T + b_h   (fp32 output)
// ---------------------------------------------------------------------------
__global__ void __launch_bounds__(256)
outproj_kernel(const unsigned short* __restrict__ attno,
               const float* __restrict__ w_h, const float* __restrict__ b_h,
               float* __restrict__ out)
{
  const int lane = threadIdx.x & 31;
  const int wave = threadIdx.x >> 5;
  const int tile = blockIdx.x * 8 + wave;
  const int rt = tile >> 5;
  const int ct = tile & 31;
  const int row = rt * 16 + (lane & 15);
  const int col = ct * 16 + (lane & 15);

  v8f acc = {};
  for (int kk = 0; kk < DM; kk += 32) {
    const int ka  = kk + (lane >> 4) * 16;
    const int kbB = kk + (lane >> 4) * 16;
    v16bf A, Bf;
#pragma unroll
    for (int j = 0; j < 8; ++j) {
      A[j]     = bits_bf(attno[row * DM + ka + j]);
      A[j + 8] = bits_bf(attno[row * DM + ka + 8 + j]);
    }
#pragma unroll
    for (int e = 0; e < 16; ++e) Bf[e] = (__bf16)w_h[col * DM + kbB + e];
    acc = wmma_bf16(A, Bf, acc);
  }
  const float bias = b_h[col];
#pragma unroll
  for (int r = 0; r < 8; ++r) {
    const int orow = rt * 16 + r + 8 * (lane >> 4);
    out[orow * DM + col] = acc[r] + bias;
  }
}

// ---------------------------------------------------------------------------
extern "C" void kernel_launch(void* const* d_in, const int* in_sizes, int n_in,
                              void* d_out, int out_size, void* d_ws, size_t ws_size,
                              hipStream_t stream) {
  const float*         q     = (const float*)d_in[0];
  const float*         k     = (const float*)d_in[1];
  const float*         edge  = (const float*)d_in[2];
  const unsigned char* pmask = (const unsigned char*)d_in[3];   // bool mask
  const float*         w_q   = (const float*)d_in[4];
  const float*         b_q   = (const float*)d_in[5];
  const float*         w_k   = (const float*)d_in[6];
  const float*         w_v   = (const float*)d_in[7];
  const float*         w_h   = (const float*)d_in[8];
  const float*         b_h   = (const float*)d_in[9];
  float*               out   = (float*)d_out;

  // Workspace: four bf16 [768 x 512] buffers (qproj, kproj, vproj-kproj, attn out)
  unsigned short* qb    = (unsigned short*)d_ws;
  unsigned short* kb    = qb + (size_t)ROWS * DM;
  unsigned short* db    = kb + (size_t)ROWS * DM;
  unsigned short* attno = db + (size_t)ROWS * DM;

  // 48 row-tiles x 32 col-tiles = 1536 wave-tiles, 8 waves/block -> 192 blocks
  proj_kernel<<<192, 256, 0, stream>>>(q, k, w_q, b_q, w_k, w_v, qb, kb, db);
  attn_kernel<<<ROWS, 256, 0, stream>>>(edge, pmask, qb, kb, db, attno);
  outproj_kernel<<<192, 256, 0, stream>>>(attno, w_h, b_h, out);
}
